// Lawformer_89704686944388
// MI455X (gfx1250) — compile-verified, matrix operand order checked by reading, not run on hardware
//
#include <hip/hip_runtime.h>
#include <hip/hip_bf16.h>
#include <math.h>

typedef __bf16 bf16_t;
typedef __attribute__((ext_vector_type(16))) __bf16 v16bf;
typedef __attribute__((ext_vector_type(8)))  float  v8f;

namespace {
constexpr int Bc  = 2;
constexpr int Sc  = 4096;
constexpr int Dc  = 768;
constexpr int Hc  = 12;
constexpr int DHc = 64;
constexpr int Wc  = 128;
constexpr int Gc  = 16;
constexpr int NBc = Sc / Wc;     // 32 query blocks
constexpr int K3c = 3 * Wc;      // 384 local window keys
constexpr int LTc = K3c + Gc;    // 400 total attention width
constexpr int LTPc = 416;        // padded to multiple of 32 for WMMA K-loop
constexpr int Mc  = Bc * Sc;     // 8192 token rows
constexpr float NEGc = -1.0e9f;
}

#define DEV static __device__ __forceinline__

// ---------------------------------------------------------------------------
// WMMA wrapper: D(16x16 f32) = A(16x32 bf16) * B(32x16 bf16) + C
// ---------------------------------------------------------------------------
DEV v8f wmma_bf16(v16bf a, v16bf b, v8f c) {
  return __builtin_amdgcn_wmma_f32_16x16x32_bf16(
      /*neg_a=*/false, a, /*neg_b=*/false, b,
      /*c_mod=*/(short)0, c, /*reuse_a=*/false, /*reuse_b=*/false);
}

DEV v16bf pack_v16(uint4 a, uint4 b) {
  union { v16bf v; uint4 u[2]; } t;
  t.u[0] = a; t.u[1] = b;
  return t.v;
}
DEV v16bf zero16() {
  union { v16bf v; uint4 u[2]; } t;
  t.u[0] = make_uint4(0u, 0u, 0u, 0u);
  t.u[1] = make_uint4(0u, 0u, 0u, 0u);
  return t.v;
}

// K-contiguous fragment (A role: line = row; B role: line = column).
// Line index = lane&15, K-half split {0-7,16-23}/{8-15,24-31} per ISA 7.12.2.
// Requires 16B alignment of p0 + m*ld + kh (all call sites satisfy this).
DEV v16bf frag_ld(const bf16_t* p0, int ld, int lane) {
  const int m  = lane & 15;
  const int kh = (lane & 16) ? 8 : 0;
  const uint4* p = (const uint4*)(p0 + (size_t)m * ld + kh);
  return pack_v16(p[0], p[2]);           // +2 uint4 = +16 bf16
}

// Same, converting f32 (probabilities in LDS / scratch) to bf16.
DEV v16bf frag_ld_f32(const float* p0, int ld, int lane) {
  const int m  = lane & 15;
  const int kh = (lane & 16) ? 8 : 0;
  const float4* p = (const float4*)(p0 + (size_t)m * ld + kh);
  const float4 x0 = p[0], x1 = p[1], y0 = p[4], y1 = p[5];
  v16bf a;
  a[0]  = (bf16_t)x0.x; a[1]  = (bf16_t)x0.y; a[2]  = (bf16_t)x0.z; a[3]  = (bf16_t)x0.w;
  a[4]  = (bf16_t)x1.x; a[5]  = (bf16_t)x1.y; a[6]  = (bf16_t)x1.z; a[7]  = (bf16_t)x1.w;
  a[8]  = (bf16_t)y0.x; a[9]  = (bf16_t)y0.y; a[10] = (bf16_t)y0.z; a[11] = (bf16_t)y0.w;
  a[12] = (bf16_t)y1.x; a[13] = (bf16_t)y1.y; a[14] = (bf16_t)y1.z; a[15] = (bf16_t)y1.w;
  return a;
}

// B fragment of K^T for the banded window: per-lane column guard, contiguous K.
DEV v16bf frag_b_keys(const bf16_t* kh_, int nblk, int col0, int k0, int lane) {
  const int nn  = lane & 15;
  const int khh = (lane & 16) ? 8 : 0;
  const int col = col0 + nn;
  const bf16_t* p = nullptr;
  if (col < K3c) {
    const int orig = nblk * Wc - Wc + col;
    if (orig >= 0 && orig < Sc) p = kh_ + (size_t)orig * DHc;
  } else if (col < LTc) {
    p = kh_ + (size_t)(col - K3c) * DHc;
  }
  if (!p) return zero16();
  const uint4* q = (const uint4*)(p + k0 + khh);
  return pack_v16(q[0], q[2]);
}

// ---------------------------------------------------------------------------
// fp32 -> bf16 conversion (plain) and transposing variant for weights
// ---------------------------------------------------------------------------
__global__ void k_cvt_bf16(const float* __restrict__ src, bf16_t* __restrict__ dst, int n) {
  const int i = blockIdx.x * blockDim.x + threadIdx.x;
  if (i < n) dst[i] = (bf16_t)src[i];
}

__global__ void k_cvt_t_bf16(const float* __restrict__ src, bf16_t* __restrict__ dst) {
  // dst[c*D + r] = src[r*D + c]   (WT layout: output-column major, K contiguous)
  const int i = blockIdx.x * blockDim.x + threadIdx.x;
  if (i < Dc * Dc) {
    const int r = i / Dc, c = i % Dc;
    dst[(size_t)c * Dc + r] = (bf16_t)src[i];
  }
}

// ---------------------------------------------------------------------------
// GEMM: C(MxD) = A(MxD bf16) @ W(DxD) + bias; W passed pre-transposed (WT).
// One wave computes a 16x32 tile (A fragment reused for two WMMAs).
// MODE 0: out bf16 (B,H,S,DH)              (q, k, kg)
// MODE 1: out bf16 (B,H,G,DH), A rows = tokens b*S+g   (qg)
// MODE 2: out f32 row-major MxD            (final projection)
// MODE 3: out bf16 (B,H,DH,S) transposed   (v, vg -> value-transposed)
// ---------------------------------------------------------------------------
template <int MODE>
DEV void gemm_body(const bf16_t* __restrict__ A, const bf16_t* __restrict__ WT,
                   const float* __restrict__ bias, void* __restrict__ out, int M,
                   float scale) {
  const int lane = threadIdx.x & 31;
  const int wid  = blockIdx.x * (blockDim.x >> 5) + (threadIdx.x >> 5);
  const int n2tiles = (Dc / 16) / 2;      // 24 (each wave: 2 adjacent N tiles)
  const int mtiles  = M / 16;
  if (wid >= mtiles * n2tiles) return;
  const int mt = wid / n2tiles, nt = wid % n2tiles;
  const int row0 = mt * 16, col0 = nt * 32;
  const int arow0 = (MODE == 1) ? (mt * Sc) : row0;   // qg: tile mt -> batch mt, tokens 0..15

  v8f acc0 = {}, acc1 = {};
  for (int k0 = 0; k0 < Dc; k0 += 32) {
    v16bf a  = frag_ld(A + (size_t)arow0 * Dc + k0, Dc, lane);
    v16bf b0 = frag_ld(WT + (size_t)col0 * Dc + k0, Dc, lane);
    v16bf b1 = frag_ld(WT + (size_t)(col0 + 16) * Dc + k0, Dc, lane);
    acc0 = wmma_bf16(a, b0, acc0);
    acc1 = wmma_bf16(a, b1, acc1);
  }

  const int nn = lane & 15;
  const int mb = (lane & 16) ? 8 : 0;
#pragma unroll
  for (int half = 0; half < 2; ++half) {
    const v8f& acc = half ? acc1 : acc0;
#pragma unroll
    for (int j = 0; j < 8; ++j) {
      const int m = row0 + mb + j;
      const int c = col0 + half * 16 + nn;
      const float v = (acc[j] + bias[c]) * scale;
      if (MODE == 2) {
        ((float*)out)[(size_t)m * Dc + c] = v;
      } else {
        const int h = c >> 6, dh = c & 63;
        if (MODE == 0) {
          const int b_ = m >> 12, s = m & 4095;
          ((bf16_t*)out)[(((size_t)(b_ * Hc + h)) * Sc + s) * DHc + dh] = (bf16_t)v;
        } else if (MODE == 1) {
          const int b_ = m >> 4, g = m & 15;
          ((bf16_t*)out)[(((size_t)(b_ * Hc + h)) * Gc + g) * DHc + dh] = (bf16_t)v;
        } else {  // MODE 3: value-transposed (B,H,DH,S)
          const int b_ = m >> 12, s = m & 4095;
          ((bf16_t*)out)[(((size_t)(b_ * Hc + h)) * DHc + dh) * Sc + s] = (bf16_t)v;
        }
      }
    }
  }
}

__global__ void k_gemm_bhsd(const bf16_t* A, const bf16_t* WT, const float* bias,
                            void* out, int M, float scale) {
  gemm_body<0>(A, WT, bias, out, M, scale);
}
__global__ void k_gemm_qg(const bf16_t* A, const bf16_t* WT, const float* bias,
                          void* out, int M, float scale) {
  gemm_body<1>(A, WT, bias, out, M, scale);
}
__global__ void k_gemm_f32(const bf16_t* A, const bf16_t* WT, const float* bias,
                           void* out, int M, float scale) {
  gemm_body<2>(A, WT, bias, out, M, scale);
}
__global__ void k_gemm_vt(const bf16_t* A, const bf16_t* WT, const float* bias,
                          void* out, int M, float scale) {
  gemm_body<3>(A, WT, bias, out, M, scale);
}

// ---------------------------------------------------------------------------
// Banded local attention + global key columns.
// Grid: B*H*NB*4 blocks of 64 threads (2 waves). Block = 32 query rows.
// q,k in (B,H,S,DH); v passed TRANSPOSED as (B,H,DH,S).
// ---------------------------------------------------------------------------
__global__ void k_local_attn(const bf16_t* __restrict__ q, const bf16_t* __restrict__ k,
                             const bf16_t* __restrict__ vt, const int* __restrict__ mask,
                             bf16_t* __restrict__ merged) {
  __shared__ alignas(16) float sc[32][LTPc];   // 32 x 416 f32 = 52 KB

  int tmp = blockIdx.x;
  const int seg = tmp & 3;  tmp >>= 2;
  const int n   = tmp % NBc; tmp /= NBc;
  const int h   = tmp % Hc;
  const int b   = tmp / Hc;

  const int lane  = threadIdx.x & 31;
  const int w     = threadIdx.x >> 5;         // 0 or 1
  const int qi0   = seg * 32;                 // first query row of this block segment
  const int rbase = w * 16;                   // this wave's row base in sc
  const int s0    = n * Wc + qi0 + rbase;     // absolute query row of wave tile

  const bf16_t* qh  = q  + ((size_t)(b * Hc + h) * Sc) * DHc;
  const bf16_t* kh  = k  + ((size_t)(b * Hc + h) * Sc) * DHc;
  const bf16_t* vth = vt + ((size_t)(b * Hc + h) * DHc) * Sc;

  // ---- Phase 1: scores = Q * K^T (16 rows per wave x 400 cols) ----
  for (int ct = 0; ct < 25; ++ct) {           // 25 tiles cover cols 0..399
    const int col0 = ct * 16;
    v8f acc = {};
#pragma unroll
    for (int k0 = 0; k0 < DHc; k0 += 32) {
      v16bf a  = frag_ld(qh + (size_t)s0 * DHc + k0, DHc, lane);
      v16bf bb = frag_b_keys(kh, n, col0, k0, lane);
      acc = wmma_bf16(a, bb, acc);
    }
    const int nn = lane & 15;
    const int mb = (lane & 16) ? 8 : 0;
#pragma unroll
    for (int j = 0; j < 8; ++j) sc[rbase + mb + j][col0 + nn] = acc[j];
  }
  __syncthreads();

  // ---- Phase 2: masking + softmax, one row per thread (rows 0..31) ----
  if (threadIdx.x < 32) {
    float* row = sc[threadIdx.x];
    const int qi = qi0 + (int)threadIdx.x;
    const int* mk = mask + (size_t)b * Sc;
    float mx = -3.4e38f;
    for (int j = 0; j < LTc; ++j) {
      float val = row[j];
      bool ok;
      if (j < K3c) {
        const int orig = n * Wc - Wc + j;
        const int rel  = j - Wc - qi;
        ok = (rel >= -Wc) && (rel <= Wc) && (orig >= Gc) && (orig < Sc) && (mk[orig] > 0);
      } else {
        ok = mk[j - K3c] > 0;
      }
      val = ok ? val : NEGc;
      row[j] = val;
      mx = fmaxf(mx, val);
    }
    float sum = 0.0f;
    for (int j = 0; j < LTc; ++j) { const float e = __expf(row[j] - mx); row[j] = e; sum += e; }
    const float inv = 1.0f / sum;
    for (int j = 0; j < LTc; ++j) row[j] *= inv;
    for (int j = LTc; j < LTPc; ++j) row[j] = 0.0f;   // zero K padding
  }
  __syncthreads();

  // ---- Phase 3: Out = P @ [V_window; V_global], 16x64 per wave ----
  // With transposed V, every 32-wide K tile is either fully in-range (one
  // vector fragment) or fully out-of-range (zero fragment; P is 0 there too,
  // since orig0 and S are both multiples of 32).
#pragma unroll
  for (int nt = 0; nt < 4; ++nt) {
    const int col0 = nt * 16;
    v8f acc = {};
    for (int k0 = 0; k0 < LTPc; k0 += 32) {
      const bf16_t* base = nullptr;
      if (k0 == K3c) {
        base = vth + (size_t)col0 * Sc;                 // global keys g = 0..15 (+16..31 * p=0)
      } else {
        const int orig0 = n * Wc - Wc + k0;
        if (orig0 >= 0 && orig0 + 32 <= Sc) base = vth + (size_t)col0 * Sc + orig0;
      }
      v16bf bb = base ? frag_ld(base, Sc, lane) : zero16();
      v16bf a  = frag_ld_f32(&sc[rbase][0] + k0, LTPc, lane);
      acc = wmma_bf16(a, bb, acc);
    }
    const int nn = lane & 15;
    const int mb = (lane & 16) ? 8 : 0;
#pragma unroll
    for (int j = 0; j < 8; ++j) {
      const int s = n * Wc + qi0 + rbase + mb + j;
      const int c = h * DHc + col0 + nn;
      merged[((size_t)b * Sc + s) * Dc + c] = (bf16_t)acc[j];
    }
  }
}

// ---------------------------------------------------------------------------
// Global-query attention: 16 rows attend over all S keys (kg / vg-transposed).
// Grid: B*H blocks of 256 threads (8 waves). Scores in global scratch.
// ---------------------------------------------------------------------------
__global__ void k_global_attn(const bf16_t* __restrict__ qg, const bf16_t* __restrict__ kg,
                              const bf16_t* __restrict__ vgt, const int* __restrict__ mask,
                              float* __restrict__ scratch, bf16_t* __restrict__ merged) {
  const int b = blockIdx.x / Hc;
  const int h = blockIdx.x % Hc;
  const int lane = threadIdx.x & 31;
  const int w    = threadIdx.x >> 5;

  const bf16_t* qh  = qg  + ((size_t)(b * Hc + h) * Gc) * DHc;
  const bf16_t* kh  = kg  + ((size_t)(b * Hc + h) * Sc) * DHc;
  const bf16_t* vth = vgt + ((size_t)(b * Hc + h) * DHc) * Sc;
  float* sg = scratch + (size_t)(b * Hc + h) * Gc * Sc;   // 16 x 4096

  // ---- Phase 1: scores 16 x S via WMMA, waves stride over column tiles ----
  for (int ct = w; ct < Sc / 16; ct += 8) {
    const int col0 = ct * 16;
    v8f acc = {};
#pragma unroll
    for (int k0 = 0; k0 < DHc; k0 += 32) {
      v16bf a  = frag_ld(qh + k0, DHc, lane);
      v16bf bb = frag_ld(kh + (size_t)col0 * DHc + k0, DHc, lane);  // column = key row, K contiguous
      acc = wmma_bf16(a, bb, acc);
    }
    const int nn = lane & 15;
    const int mb = (lane & 16) ? 8 : 0;
#pragma unroll
    for (int j = 0; j < 8; ++j) sg[(size_t)(mb + j) * Sc + col0 + nn] = acc[j];
  }
  __syncthreads();

  // ---- Phase 2: masked softmax per row (threads 0..15) ----
  if (threadIdx.x < Gc) {
    float* row = sg + (size_t)threadIdx.x * Sc;
    const int* mk = mask + (size_t)b * Sc;
    float mx = -3.4e38f;
    for (int j = 0; j < Sc; ++j) {
      const float vv = (mk[j] > 0) ? row[j] : NEGc;
      row[j] = vv;
      mx = fmaxf(mx, vv);
    }
    float sum = 0.0f;
    for (int j = 0; j < Sc; ++j) { const float e = __expf(row[j] - mx); row[j] = e; sum += e; }
    const float inv = 1.0f / sum;
    for (int j = 0; j < Sc; ++j) row[j] *= inv;
  }
  __syncthreads();

  // ---- Phase 3: Out(16x64) = P(16xS) @ Vg(Sx64); waves 0..3 ----
  if (w < 4) {
    const int col0 = w * 16;
    v8f acc = {};
    for (int k0 = 0; k0 < Sc; k0 += 32) {
      v16bf a  = frag_ld_f32(sg + k0, Sc, lane);
      v16bf bb = frag_ld(vth + (size_t)col0 * Sc + k0, Sc, lane);   // K contiguous via transpose
      acc = wmma_bf16(a, bb, acc);
    }
    const int nn = lane & 15;
    const int mb = (lane & 16) ? 8 : 0;
#pragma unroll
    for (int j = 0; j < 8; ++j) {
      const int g = mb + j;
      const int c = h * DHc + col0 + nn;
      merged[((size_t)b * Sc + g) * Dc + c] = (bf16_t)acc[j];   // overwrite rows < G
    }
  }
}

// ---------------------------------------------------------------------------
// Host launcher
// ---------------------------------------------------------------------------
extern "C" void kernel_launch(void* const* d_in, const int* in_sizes, int n_in,
                              void* d_out, int out_size, void* d_ws, size_t ws_size,
                              hipStream_t stream) {
  (void)in_sizes; (void)n_in; (void)out_size; (void)ws_size;

  const float* x    = (const float*)d_in[0];
  const int*   mask = (const int*)d_in[1];
  const float* Wq   = (const float*)d_in[2];  const float* bq  = (const float*)d_in[3];
  const float* Wk   = (const float*)d_in[4];  const float* bk  = (const float*)d_in[5];
  const float* Wv   = (const float*)d_in[6];  const float* bv  = (const float*)d_in[7];
  const float* Wqg  = (const float*)d_in[8];  const float* bqg = (const float*)d_in[9];
  const float* Wkg  = (const float*)d_in[10]; const float* bkg = (const float*)d_in[11];
  const float* Wvg  = (const float*)d_in[12]; const float* bvg = (const float*)d_in[13];
  const float* Wo   = (const float*)d_in[14]; const float* bo  = (const float*)d_in[15];

  char* ws = (char*)d_ws;
  size_t off = 0;
  auto carve = [&](size_t bytes) -> char* {
    char* p = ws + off;
    off += (bytes + 255) & ~(size_t)255;
    return p;
  };

  const size_t nTok  = (size_t)Mc * Dc;
  const size_t nHead = (size_t)Bc * Hc * Sc * DHc;

  bf16_t* xb   = (bf16_t*)carve(nTok * 2);
  bf16_t* wqb  = (bf16_t*)carve((size_t)Dc * Dc * 2);   // all weights stored transposed
  bf16_t* wkb  = (bf16_t*)carve((size_t)Dc * Dc * 2);
  bf16_t* wvb  = (bf16_t*)carve((size_t)Dc * Dc * 2);
  bf16_t* wqgb = (bf16_t*)carve((size_t)Dc * Dc * 2);
  bf16_t* wkgb = (bf16_t*)carve((size_t)Dc * Dc * 2);
  bf16_t* wvgb = (bf16_t*)carve((size_t)Dc * Dc * 2);
  bf16_t* wob  = (bf16_t*)carve((size_t)Dc * Dc * 2);
  bf16_t* qb   = (bf16_t*)carve(nHead * 2);             // (B,H,S,DH)
  bf16_t* kb   = (bf16_t*)carve(nHead * 2);             // (B,H,S,DH)
  bf16_t* vtb  = (bf16_t*)carve(nHead * 2);             // (B,H,DH,S) transposed
  bf16_t* kgb  = (bf16_t*)carve(nHead * 2);             // (B,H,S,DH)
  bf16_t* vgtb = (bf16_t*)carve(nHead * 2);             // (B,H,DH,S) transposed
  bf16_t* qgb  = (bf16_t*)carve((size_t)Bc * Hc * Gc * DHc * 2);
  bf16_t* mg   = (bf16_t*)carve(nTok * 2);
  float*  sg   = (float*)carve((size_t)Bc * Hc * Gc * Sc * 4);

  // fp32 -> bf16 conversions (weights transposed so GEMM B is K-contiguous)
  k_cvt_bf16<<<((int)nTok + 255) / 256, 256, 0, stream>>>(x, xb, (int)nTok);
  const int wblocks = (Dc * Dc + 255) / 256;
  k_cvt_t_bf16<<<wblocks, 256, 0, stream>>>(Wq,  wqb);
  k_cvt_t_bf16<<<wblocks, 256, 0, stream>>>(Wk,  wkb);
  k_cvt_t_bf16<<<wblocks, 256, 0, stream>>>(Wv,  wvb);
  k_cvt_t_bf16<<<wblocks, 256, 0, stream>>>(Wqg, wqgb);
  k_cvt_t_bf16<<<wblocks, 256, 0, stream>>>(Wkg, wkgb);
  k_cvt_t_bf16<<<wblocks, 256, 0, stream>>>(Wvg, wvgb);
  k_cvt_t_bf16<<<wblocks, 256, 0, stream>>>(Wo,  wob);

  const float scale = 0.125f;                         // 1/sqrt(64)
  const int waves   = (Mc / 16) * ((Dc / 16) / 2);    // 12288 (16x32 tiles)
  const int gblocks = (waves + 7) / 8;                // 1536 blocks of 8 waves

  k_gemm_bhsd<<<gblocks, 256, 0, stream>>>(xb, wqb,  bq,  qb,   Mc, scale);
  k_gemm_bhsd<<<gblocks, 256, 0, stream>>>(xb, wkb,  bk,  kb,   Mc, 1.0f);
  k_gemm_vt  <<<gblocks, 256, 0, stream>>>(xb, wvb,  bv,  vtb,  Mc, 1.0f);
  k_gemm_bhsd<<<gblocks, 256, 0, stream>>>(xb, wkgb, bkg, kgb,  Mc, 1.0f);
  k_gemm_vt  <<<gblocks, 256, 0, stream>>>(xb, wvgb, bvg, vgtb, Mc, 1.0f);

  const int qgwaves = (32 / 16) * ((Dc / 16) / 2);    // 48
  k_gemm_qg<<<(qgwaves + 7) / 8, 256, 0, stream>>>(xb, wqgb, bqg, qgb, 32, scale);

  k_local_attn<<<Bc * Hc * NBc * 4, 64, 0, stream>>>(qb, kb, vtb, mask, mg);
  k_global_attn<<<Bc * Hc, 256, 0, stream>>>(qgb, kgb, vgtb, mask, sg, mg);

  k_gemm_f32<<<gblocks, 256, 0, stream>>>(mg, wob, bo, d_out, Mc, 1.0f);
}
